// BahdanauAttention_26834955665671
// MI455X (gfx1250) — matrix operand order repaired and checked by reading, not compiled, not run on hardware
//
#include <hip/hip_runtime.h>
#include <hip/hip_bf16.h>

#define B_ 32
#define T_ 8192
#define D_ 256
#define U_ 128
#define LDS_STRIDE 264   // bf16 elems per W1T row: 528B -> 16B aligned, lane bank stride 4

typedef __attribute__((ext_vector_type(16))) __bf16 v16bf;
typedef __attribute__((ext_vector_type(8)))  float  v8f;

// Native CDNA5 tanh (V_TANH_F32) when available; libm fallback otherwise.
__device__ __forceinline__ float fast_tanh(float x) {
#if defined(__has_builtin)
#if __has_builtin(__builtin_amdgcn_tanhf)
  return __builtin_amdgcn_tanhf(x);
#else
  return tanhf(x);
#endif
#else
  return tanhf(x);
#endif
}

// ---------------- K0: h_q[b,u] = relu(query[b,:] @ W2[:,u] + b2[u]) ----------------
__global__ __launch_bounds__(256) void hq_kernel(const float* __restrict__ q,
                                                 const float* __restrict__ W2,
                                                 const float* __restrict__ b2,
                                                 float* __restrict__ hq) {
  int idx = blockIdx.x * blockDim.x + threadIdx.x;   // b*U + u
  int b = idx >> 7, u = idx & (U_ - 1);
  float acc = b2[u];
  const float* qr = q + (size_t)b * D_;
  for (int d = 0; d < D_; ++d) acc += qr[d] * W2[d * U_ + u];
  hq[idx] = fmaxf(acc, 0.f);
}

// ---------------- K1: raw scores via bf16 WMMA ----------------
// Each wave: 16 rows of t x all 128 u, K=256 in 8 steps of 32.
__global__ __launch_bounds__(256) void score_kernel(const float* __restrict__ values,
                                                    const float* __restrict__ W1,
                                                    const float* __restrict__ b1,
                                                    const float* __restrict__ hq,
                                                    const float* __restrict__ Vk,
                                                    const float* __restrict__ Vb,
                                                    float* __restrict__ scores) {
  __shared__ __bf16 w1t[U_ * LDS_STRIDE];            // W1 transposed, bf16 (~66 KB of 320 KB)
  int tid = threadIdx.x;

  // Stage W1^T into LDS (coalesced global read, padded LDS write)
  for (int idx = tid; idx < D_ * U_; idx += 256) {
    int d = idx >> 7, u = idx & (U_ - 1);
    w1t[u * LDS_STRIDE + d] = (__bf16)W1[idx];
  }
  __syncthreads();

  int b    = blockIdx.x >> 6;                        // 64 blocks per batch
  int wave = tid >> 5;
  int lane = tid & 31;
  int t0   = (((blockIdx.x & 63) << 3) + wave) << 4; // 8 waves/block * 16 rows
  int n    = lane & 15;
  int hi   = lane >> 4;                              // half-wave select

  // Per-lane epilogue constants for u = ut*16 + n
  float hql[8], vkl[8], b1l[8];
#pragma unroll
  for (int ut = 0; ut < 8; ++ut) {
    int u = ut * 16 + n;
    hql[ut] = hq[b * U_ + u];
    vkl[ut] = Vk[u];
    b1l[ut] = b1[u];
  }

  // A-fragment source: this lane's t-row (A 16x32 bf16, M = lane&15,
  // lanes 0-15 carry K 0-15, lanes 16-31 carry K 16-31)
  const float* rowA = values + ((size_t)b * T_ + t0 + n) * D_;
  int kbA = hi ? 16 : 0;

  v8f c[8];
#pragma unroll
  for (int ut = 0; ut < 8; ++ut) c[ut] = (v8f){0.f, 0.f, 0.f, 0.f, 0.f, 0.f, 0.f, 0.f};

  for (int ks = 0; ks < 8; ++ks) {
    int k0 = ks * 32;
    v16bf a;
#pragma unroll
    for (int i = 0; i < 8; ++i) {
      a[i]     = (__bf16)rowA[k0 + kbA + i];
      a[8 + i] = (__bf16)rowA[k0 + kbA + 8 + i];
    }
#pragma unroll
    for (int ut = 0; ut < 8; ++ut) {
      // B 32x16 bf16: column N = lane&15; lanes 0-15 carry K 0-15, lanes 16-31 K 16-31
      const __bf16* rb = &w1t[(ut * 16 + n) * LDS_STRIDE + k0 + (hi ? 16 : 0)];
      v16bf bb;
#pragma unroll
      for (int i = 0; i < 16; ++i) bb[i] = rb[i];
      c[ut] = __builtin_amdgcn_wmma_f32_16x16x32_bf16(false, a, false, bb,
                                                      (short)0, c[ut], false, false);
    }
  }

  // Epilogue: relu(hv + b1) + hq -> tanh -> * Vk, reduce over u.
  // f32 C/D layout: VGPR r = row (r + hi*8), column = lane&15.
  float acc[8];
#pragma unroll
  for (int r = 0; r < 8; ++r) acc[r] = 0.f;
#pragma unroll
  for (int ut = 0; ut < 8; ++ut) {
#pragma unroll
    for (int r = 0; r < 8; ++r) {
      float h = fmaxf(c[ut][r] + b1l[ut], 0.f) + hql[ut];
      acc[r] += fast_tanh(h) * vkl[ut];
    }
  }
  // Butterfly reduction across the 16 lanes of each half-wave
#pragma unroll
  for (int m = 1; m < 16; m <<= 1)
#pragma unroll
    for (int r = 0; r < 8; ++r) acc[r] += __shfl_xor(acc[r], m, 32);

  if (n == 0) {
    float vb = Vb[0];
    int trow = t0 + hi * 8;
#pragma unroll
    for (int r = 0; r < 8; ++r) scores[(size_t)b * T_ + trow + r] = acc[r] + vb;
  }
}

// ---------------- K2: softmax over T, in place ----------------
__global__ __launch_bounds__(256) void softmax_kernel(float* __restrict__ attn) {
  __shared__ float red[256];
  int b = blockIdx.x, tid = threadIdx.x;
  float* s = attn + (size_t)b * T_;
  float loc[32];
  float mx = -3.4e38f;
#pragma unroll
  for (int i = 0; i < 32; ++i) { loc[i] = s[tid + i * 256]; mx = fmaxf(mx, loc[i]); }
  red[tid] = mx; __syncthreads();
  for (int o = 128; o > 0; o >>= 1) {
    if (tid < o) red[tid] = fmaxf(red[tid], red[tid + o]);
    __syncthreads();
  }
  mx = red[0]; __syncthreads();
  float sum = 0.f;
#pragma unroll
  for (int i = 0; i < 32; ++i) { loc[i] = __expf(loc[i] - mx); sum += loc[i]; }
  red[tid] = sum; __syncthreads();
  for (int o = 128; o > 0; o >>= 1) {
    if (tid < o) red[tid] += red[tid + o];
    __syncthreads();
  }
  float inv = 1.f / red[0];
#pragma unroll
  for (int i = 0; i < 32; ++i) s[tid + i * 256] = loc[i] * inv;
}

// ---------------- K3: partial weighted sums over 128-t chunks ----------------
__global__ __launch_bounds__(256) void context_partial_kernel(const float* __restrict__ values,
                                                              const float* __restrict__ attn,
                                                              float* __restrict__ part) {
  int b = blockIdx.y, tc = blockIdx.x, d = threadIdx.x;
  const float* vb = values + ((size_t)b * T_ + (size_t)tc * 128) * D_;
  const float* ab = attn + (size_t)b * T_ + (size_t)tc * 128;
  float acc = 0.f;
  for (int t = 0; t < 128; ++t) acc += ab[t] * vb[(size_t)t * D_ + d];
  part[((size_t)b * 64 + tc) * D_ + d] = acc;
}

// ---------------- K4: deterministic tree reduce of partials ----------------
__global__ __launch_bounds__(256) void context_reduce_kernel(const float* __restrict__ part,
                                                             float* __restrict__ ctx) {
  int idx = blockIdx.x * blockDim.x + threadIdx.x;   // b*D + d
  int b = idx >> 8, d = idx & (D_ - 1);
  float acc = 0.f;
  for (int tc = 0; tc < 64; ++tc) acc += part[((size_t)b * 64 + tc) * D_ + d];
  ctx[idx] = acc;
}

extern "C" void kernel_launch(void* const* d_in, const int* in_sizes, int n_in,
                              void* d_out, int out_size, void* d_ws, size_t ws_size,
                              hipStream_t stream) {
  const float* query  = (const float*)d_in[0];
  const float* values = (const float*)d_in[1];
  const float* W1k    = (const float*)d_in[2];
  const float* W1b    = (const float*)d_in[3];
  const float* W2k    = (const float*)d_in[4];
  const float* W2b    = (const float*)d_in[5];
  const float* Vkn    = (const float*)d_in[6];
  const float* Vbs    = (const float*)d_in[7];

  float* ctx  = (float*)d_out;                 // [B, D]
  float* attn = (float*)d_out + B_ * D_;       // [B, T] (scores in place, then softmax)
  float* hq   = (float*)d_ws;                  // [B, U]
  float* part = hq + B_ * U_;                  // [B, 64, D]

  hq_kernel<<<(B_ * U_) / 256, 256, 0, stream>>>(query, W2k, W2b, hq);
  score_kernel<<<B_ * 64, 256, 0, stream>>>(values, W1k, W1b, hq, Vkn, Vbs, attn);
  softmax_kernel<<<B_, 256, 0, stream>>>(attn);
  context_partial_kernel<<<dim3(64, B_), 256, 0, stream>>>(values, attn, part);
  context_reduce_kernel<<<(B_ * D_) / 256, 256, 0, stream>>>(part, ctx);
}
